// GATLayer_81767587381920
// MI455X (gfx1250) — matrix operand order repaired and checked by reading, not compiled, not run on hardware
//
#include <hip/hip_runtime.h>

typedef __attribute__((ext_vector_type(16))) __bf16 v16bf;
typedef __attribute__((ext_vector_type(8)))  float  v8f;

#define D 128                     // feature dim (in == out == edge)
#define FRAG_ELEMS (8*8*32*16)    // 32768 bf16 per weight table (64 KB)

// ---------------- prep: zero the h_neigh accumulator (lives in d_out) -------
__global__ void k_zero4(float4* __restrict__ p, int n4) {
  int i = blockIdx.x * blockDim.x + threadIdx.x;
  if (i < n4) p[i] = make_float4(0.f, 0.f, 0.f, 0.f);
}

// ---------------- prep: build bf16 B-fragment tables for both weights -------
// Table layout: [kc(8)][nt(8)][lane(32)][i(16)]  (bf16)
//   n = nt*16 + (lane&15); khalf = lane>>4; k = kc*32 + khalf*8 + (i&7) + (i>>3)*16
// W is [128 out][256 in] row-major; B[k][n] = W[n][k].
__global__ void k_build_bfrag(const float* __restrict__ Wmsg,
                              const float* __restrict__ Wapp,
                              __bf16* __restrict__ frag) {
  int idx = blockIdx.x * 256 + threadIdx.x;        // 0 .. 2*FRAG_ELEMS-1
  const float* W = (idx & FRAG_ELEMS) ? Wapp : Wmsg;
  int rem  = idx & (FRAG_ELEMS - 1);
  int i    = rem & 15;
  int lane = (rem >> 4) & 31;
  int fid  = rem >> 9;                             // kc*8 + nt
  int nt   = fid & 7;
  int kc   = fid >> 3;
  int k    = kc * 32 + (lane >> 4) * 8 + (i & 7) + (i >> 3) * 16;
  int n    = nt * 16 + (lane & 15);
  frag[idx] = (__bf16)W[n * 256 + k];
}

// Build one 16x32 bf16 A fragment per the CDNA5 16-bit A layout:
//  half h==0 (lanes 0-15):  elems 0-7 -> K 0-7,  elems 8-15 -> K 16-23
//  half h==1 (lanes 16-31): elems 0-7 -> K 8-15, elems 8-15 -> K 24-31
__device__ __forceinline__ v16bf load_a_frag(const float* __restrict__ base, int h) {
  const float4* p0 = (const float4*)(base + h * 8);
  const float4* p1 = (const float4*)(base + 16 + h * 8);
  float4 f0 = p0[0], f1 = p0[1], f2 = p1[0], f3 = p1[1];
  v16bf a;
  a[0]  = (__bf16)f0.x; a[1]  = (__bf16)f0.y; a[2]  = (__bf16)f0.z; a[3]  = (__bf16)f0.w;
  a[4]  = (__bf16)f1.x; a[5]  = (__bf16)f1.y; a[6]  = (__bf16)f1.z; a[7]  = (__bf16)f1.w;
  a[8]  = (__bf16)f2.x; a[9]  = (__bf16)f2.y; a[10] = (__bf16)f2.z; a[11] = (__bf16)f2.w;
  a[12] = (__bf16)f3.x; a[13] = (__bf16)f3.y; a[14] = (__bf16)f3.z; a[15] = (__bf16)f3.w;
  return a;
}

// ---------------- edge kernel: z = Wmsg·cat(h_src, ef) + b; scatter-add -----
// Each wave processes TWO 16-edge tiles: every B fragment loaded from LDS
// feeds two back-to-back WMMAs, halving ds-wait stalls per matrix op.
__global__ __launch_bounds__(256) void k_gat_edge(
    const float* __restrict__ nfeats, const float* __restrict__ efeats,
    const int* __restrict__ src, const int* __restrict__ dst,
    const float* __restrict__ bias, const __bf16* __restrict__ Bfrag,
    float* __restrict__ hneigh, int numTiles) {
  __shared__ __align__(32) __bf16 lB[FRAG_ELEMS];
  {
    const uint4* gs = (const uint4*)Bfrag;
    uint4* ls = (uint4*)lB;
    for (int i = threadIdx.x; i < FRAG_ELEMS / 8; i += 256) ls[i] = gs[i];
  }
  __syncthreads();

  int wave = threadIdx.x >> 5;
  int lane = threadIdx.x & 31;
  int pair = blockIdx.x * 8 + wave;
  int t0 = pair * 2;
  if (t0 >= numTiles) return;            // uniform per wave: EXEC stays all-1s
  bool has1 = (t0 + 1) < numTiles;       // uniform

  int r = lane & 15, h = lane >> 4;
  int e0 = t0 * 16;
  int e1 = e0 + 16;

  int se0 = src[e0 + r];
  int se1 = has1 ? src[e1 + r] : se0;
  const float* sRow0 = nfeats + (long)se0 * D;
  const float* sRow1 = nfeats + (long)se1 * D;
  const float* eRow0 = efeats + (long)(e0 + r) * D;
  const float* eRow1 = has1 ? (efeats + (long)(e1 + r) * D) : eRow0;

  v8f acc0[8], acc1[8];
  #pragma unroll
  for (int nt = 0; nt < 8; ++nt) {
    float bm = bias[nt * 16 + r];        // C layout: lane -> column n
    #pragma unroll
    for (int v = 0; v < 8; ++v) { acc0[nt][v] = bm; acc1[nt][v] = bm; }
  }

  #pragma unroll
  for (int kc = 0; kc < 8; ++kc) {
    const float* b0 = (kc < 4) ? (sRow0 + kc * 32) : (eRow0 + (kc * 32 - 128));
    const float* b1 = (kc < 4) ? (sRow1 + kc * 32) : (eRow1 + (kc * 32 - 128));
    v16bf a0 = load_a_frag(b0, h);
    v16bf a1 = load_a_frag(b1, h);
    #pragma unroll
    for (int nt = 0; nt < 8; ++nt) {
      v16bf b = *(const v16bf*)(&lB[(((kc << 3) | nt) * 32 + lane) * 16]);
      acc0[nt] = __builtin_amdgcn_wmma_f32_16x16x32_bf16(
          false, a0, false, b, (short)0, acc0[nt], false, false);
      acc1[nt] = __builtin_amdgcn_wmma_f32_16x16x32_bf16(
          false, a1, false, b, (short)0, acc1[nt], false, false);
    }
  }

  // D layout: lanes 0-15 VGPR v -> M=v; lanes 16-31 -> M=v+8; N = lane&15
  int dn0[8];
  #pragma unroll
  for (int v = 0; v < 8; ++v) dn0[v] = dst[e0 + h * 8 + v];
  #pragma unroll
  for (int nt = 0; nt < 8; ++nt) {
    int n = nt * 16 + r;
    #pragma unroll
    for (int v = 0; v < 8; ++v)
      atomicAdd(&hneigh[(long)dn0[v] * D + n], acc0[nt][v]);
  }
  if (has1) {
    int dn1[8];
    #pragma unroll
    for (int v = 0; v < 8; ++v) dn1[v] = dst[e1 + h * 8 + v];
    #pragma unroll
    for (int nt = 0; nt < 8; ++nt) {
      int n = nt * 16 + r;
      #pragma unroll
      for (int v = 0; v < 8; ++v)
        atomicAdd(&hneigh[(long)dn1[v] * D + n], acc1[nt][v]);
    }
  }
}

// -------- apply kernel: out = relu(Wapp·cat(nfeats, h_neigh) + b), in-place --
__global__ __launch_bounds__(256) void k_gat_apply(
    const float* __restrict__ nfeats, const float* __restrict__ hneigh,
    const float* __restrict__ bias, const __bf16* __restrict__ Bfrag,
    float* __restrict__ out, int numTiles) {
  __shared__ __align__(32) __bf16 lB[FRAG_ELEMS];
  {
    const uint4* gs = (const uint4*)Bfrag;
    uint4* ls = (uint4*)lB;
    for (int i = threadIdx.x; i < FRAG_ELEMS / 8; i += 256) ls[i] = gs[i];
  }
  __syncthreads();

  int wave = threadIdx.x >> 5;
  int lane = threadIdx.x & 31;
  int tile = blockIdx.x * 8 + wave;
  if (tile >= numTiles) return;

  int r = lane & 15, h = lane >> 4;
  int n0 = tile * 16;
  const float* nfRow = nfeats + (long)(n0 + r) * D;
  const float* hnRow = hneigh + (long)(n0 + r) * D;

  v8f acc[8];
  #pragma unroll
  for (int nt = 0; nt < 8; ++nt) {
    float bm = bias[nt * 16 + r];
    #pragma unroll
    for (int v = 0; v < 8; ++v) acc[nt][v] = bm;
  }

  #pragma unroll
  for (int kc = 0; kc < 8; ++kc) {
    const float* base = (kc < 4) ? (nfRow + kc * 32) : (hnRow + (kc * 32 - 128));
    v16bf a = load_a_frag(base, h);
    #pragma unroll
    for (int nt = 0; nt < 8; ++nt) {
      v16bf b = *(const v16bf*)(&lB[(((kc << 3) | nt) * 32 + lane) * 16]);
      acc[nt] = __builtin_amdgcn_wmma_f32_16x16x32_bf16(
          false, a, false, b, (short)0, acc[nt], false, false);
    }
  }

  // all loads above complete before any store issues in this wave -> safe in-place
  #pragma unroll
  for (int nt = 0; nt < 8; ++nt) {
    int n = nt * 16 + r;
    #pragma unroll
    for (int v = 0; v < 8; ++v)
      out[(long)(n0 + h * 8 + v) * D + n] = fmaxf(acc[nt][v], 0.f);
  }
}

extern "C" void kernel_launch(void* const* d_in, const int* in_sizes, int n_in,
                              void* d_out, int out_size, void* d_ws, size_t ws_size,
                              hipStream_t stream) {
  const float* nfeats = (const float*)d_in[0];
  const float* efeats = (const float*)d_in[1];
  const int*   src    = (const int*)d_in[2];
  const int*   dst    = (const int*)d_in[3];
  const float* Wmsg   = (const float*)d_in[4];
  const float* bmsg   = (const float*)d_in[5];
  // d_in[6] = attn_w: softmax over a singleton axis makes attention a no-op.
  const float* Wapp   = (const float*)d_in[7];
  const float* bapp   = (const float*)d_in[8];
  float* out = (float*)d_out;

  int E = in_sizes[2];
  int N = in_sizes[0] / D;

  __bf16* frag = (__bf16*)d_ws;          // 2 * 64 KB weight fragment tables

  // 1) zero h_neigh accumulator (stored in d_out)
  int n4 = (N * D) / 4;
  k_zero4<<<(n4 + 255) / 256, 256, 0, stream>>>((float4*)out, n4);

  // 2) repack both weight matrices into bf16 WMMA B-fragment order
  k_build_bfrag<<<(2 * FRAG_ELEMS) / 256, 256, 0, stream>>>(Wmsg, Wapp, frag);

  // 3) edge GEMM + scatter-add into h_neigh (2 tiles per wave)
  int eTiles = E / 16;                   // E = 800000 -> 50000 tiles
  int ePairs = (eTiles + 1) / 2;         // 25000 tile pairs
  k_gat_edge<<<(ePairs + 7) / 8, 256, 0, stream>>>(
      nfeats, efeats, src, dst, bmsg, frag, out, eTiles);

  // 4) apply GEMM + bias + ReLU (reads h_neigh from d_out, writes d_out)
  int nTiles = N / 16;                   // N = 50000 -> 3125 tiles
  k_gat_apply<<<(nTiles + 7) / 8, 256, 0, stream>>>(
      nfeats, out, bapp, frag + FRAG_ELEMS, out, nTiles);
}
